// BinaryPathEncoder_77232101916966
// MI455X (gfx1250) — matrix operand order repaired and checked by reading, not compiled, not run on hardware
//
#include <hip/hip_runtime.h>
#include <hip/hip_bf16.h>

// ---------------------------------------------------------------------------
// BinaryPathEncoder for MI455X (gfx1250, wave32, WMMA)
// Strategy: f16x3 split-precision chained GEMMs via v_wmma_f32_16x16x32_f16.
// ---------------------------------------------------------------------------

typedef __attribute__((ext_vector_type(16))) _Float16 v16h;
typedef __attribute__((ext_vector_type(8)))  _Float16 h8;
typedef __attribute__((ext_vector_type(8)))  float    v8f;

#define D 256
#define DD 65536           // 256*256
#define LDS_STRIDE 264     // halves per row: 256 + 8 pad (16B) -> 4-bank rotate/row

// ------------------------- expm prep kernels (tiny cost) --------------------

// A = (prim - prim^T) * 2^-6   (scaling for scaling-and-squaring)
__global__ void k_skew(const float* __restrict__ prim, float* __restrict__ A) {
    int idx = blockIdx.x * 256 + threadIdx.x;          // 131072 total
    int b = idx >> 16, r = (idx >> 8) & 255, c = idx & 255;
    A[idx] = (prim[(b << 16) + (r << 8) + c] - prim[(b << 16) + (c << 8) + r]) * 0.015625f;
}

// E = I + A
__global__ void k_addI(const float* __restrict__ A, float* __restrict__ E) {
    int idx = blockIdx.x * 256 + threadIdx.x;
    int r = (idx >> 8) & 255, c = idx & 255;
    E[idx] = A[idx] + ((r == c) ? 1.0f : 0.0f);
}

// Tout = (Tin @ A) * invk ; E += Tout        grid (256,2) x 256
__global__ void k_term(const float* __restrict__ Tin, const float* __restrict__ A,
                       float* __restrict__ Tout, float* __restrict__ E, float invk) {
    int b = blockIdx.y, i = blockIdx.x, j = threadIdx.x;
    const float* tr = Tin + (b << 16) + (i << 8);
    const float* ac = A + (b << 16) + j;
    float acc = 0.0f;
    for (int kk = 0; kk < D; ++kk) acc = fmaf(tr[kk], ac[kk << 8], acc);
    acc *= invk;
    int o = (b << 16) + (i << 8) + j;
    Tout[o] = acc;
    E[o] += acc;
}

// Eout = Ein @ Ein            grid (256,2) x 256
__global__ void k_sq(const float* __restrict__ Ein, float* __restrict__ Eout) {
    int b = blockIdx.y, i = blockIdx.x, j = threadIdx.x;
    const float* tr = Ein + (b << 16) + (i << 8);
    const float* ac = Ein + (b << 16) + j;
    float acc = 0.0f;
    for (int kk = 0; kk < D; ++kk) acc = fmaf(tr[kk], ac[kk << 8], acc);
    Eout[(b << 16) + (i << 8) + j] = acc;
}

// M = P^T, split into f16 hi/lo; store row-major (LDS seeding) and packed into
// the WMMA B-fragment layout:  packed[b][kc][tj][lane][h], lane = (kin>=16)<<4 | (j&15),
// K = 32*kc + (lane>=16?16:0) + h, N = 16*tj + (lane&15).
__global__ void k_pack(const float* __restrict__ P,
                       _Float16* __restrict__ MhRM, _Float16* __restrict__ MlRM,
                       _Float16* __restrict__ MhPK, _Float16* __restrict__ MlPK) {
    int idx = blockIdx.x * 256 + threadIdx.x;          // b, i (=K row of M), j (=N col)
    int b = idx >> 16, i = (idx >> 8) & 255, j = idx & 255;
    float m = P[(b << 16) + (j << 8) + i];             // transpose: M = P^T
    _Float16 hi = (_Float16)m;
    _Float16 lo = (_Float16)(m - (float)hi);
    MhRM[idx] = hi;
    MlRM[idx] = lo;
    int kc = i >> 5, kin = i & 31;
    int lane = ((kin >> 4) << 4) | (j & 15);
    int h = kin & 15, tj = j >> 4;
    int off = (b << 16) + ((((kc << 4) + tj) * 32 + lane) << 4) + h;
    MhPK[off] = hi;
    MlPK[off] = lo;
}

// --------------------------- main chain kernel ------------------------------
// One block per position. 1024 threads = 32 wave32. Wave w owns output tiles
// (ti = w>>1, cols (w&1)*128..+127) -> 8 C tiles of 16x16 f32 in VGPRs.
// R kept in LDS as f16 hi/lo with padded rows; each step R <- R @ M_bit via
// 3 WMMAs per K-chunk (hi*hi + hi*lo + lo*hi).
__global__ __launch_bounds__(1024, 1)
void k_chain(const int* __restrict__ unique,
             const _Float16* __restrict__ MhRM, const _Float16* __restrict__ MlRM,
             const _Float16* __restrict__ MhPK, const _Float16* __restrict__ MlPK,
             float* __restrict__ out) {
    __shared__ _Float16 lds[2 * D * LDS_STRIDE];       // 264 KB
    _Float16* Rh = lds;
    _Float16* Rl = lds + D * LDS_STRIDE;

    const int tid = threadIdx.x;
    const unsigned p = (unsigned)unique[blockIdx.x];
    const int klen = 31 - __clz((int)p);               // path length (p >= 1)
    float* outp = out + (size_t)blockIdx.x * DD;

    if (klen == 0) {                                   // p == 1: identity output
        for (int idx = tid; idx < DD; idx += 1024) {
            int r = idx >> 8, c = idx & 255;
            outp[idx] = (r == c) ? 1.0f : 0.0f;
        }
        return;
    }

    // Seed R = M[bit0] (first factor of the chain: I @ M_b0 = M_b0)
    {
        const int b0 = p & 1;
        const _Float16* sH = MhRM + (b0 << 16);
        const _Float16* sL = MlRM + (b0 << 16);
        for (int chunk = tid; chunk < 8192; chunk += 1024) {
            int row = chunk >> 5, c8 = (chunk & 31) << 3;
            *(h8*)&Rh[row * LDS_STRIDE + c8] = *(const h8*)&sH[(row << 8) + c8];
            *(h8*)&Rl[row * LDS_STRIDE + c8] = *(const h8*)&sL[(row << 8) + c8];
        }
    }
    __syncthreads();

    const int wave = tid >> 5, lane = tid & 31;
    const int ti = wave >> 1, cg = wave & 1;
    const int mrow = lane & 15;        // M/N index within a 16x16 tile
    const int laneHi = lane >> 4;      // selects K/row halves per ISA layouts
    const v8f vzero = {};

    if (klen > 1) {
        v8f C[8];
        for (int s = 1; s < klen; ++s) {
            const int b = (p >> s) & 1;
            const _Float16* Bh = MhPK + (b << 16);
            const _Float16* Bl = MlPK + (b << 16);
#pragma unroll
            for (int j = 0; j < 8; ++j) C[j] = vzero;

            for (int kc = 0; kc < 8; ++kc) {
                // A fragments from LDS (contiguous K order: lane half h holds K = 16h..16h+15, matching B packing)
                v16h Ah, Al;
                const int abase = (ti * 16 + mrow) * LDS_STRIDE + kc * 32 + laneHi * 16;
                ((h8*)&Ah)[0] = *(h8*)&Rh[abase];
                ((h8*)&Ah)[1] = *(h8*)&Rh[abase + 8];
                ((h8*)&Al)[0] = *(h8*)&Rl[abase];
                ((h8*)&Al)[1] = *(h8*)&Rl[abase + 8];
#pragma unroll
                for (int j = 0; j < 8; ++j) {
                    const int tj = cg * 8 + j;
                    const int boff = ((((kc << 4) + tj) * 32 + lane) << 4);
                    v16h BhF, BlF;
                    ((h8*)&BhF)[0] = *(const h8*)&Bh[boff];
                    ((h8*)&BhF)[1] = *(const h8*)&Bh[boff + 8];
                    ((h8*)&BlF)[0] = *(const h8*)&Bl[boff];
                    ((h8*)&BlF)[1] = *(const h8*)&Bl[boff + 8];
                    C[j] = __builtin_amdgcn_wmma_f32_16x16x32_f16(
                        false, Ah, false, BhF, (short)0, C[j], false, false);
                    C[j] = __builtin_amdgcn_wmma_f32_16x16x32_f16(
                        false, Ah, false, BlF, (short)0, C[j], false, false);
                    C[j] = __builtin_amdgcn_wmma_f32_16x16x32_f16(
                        false, Al, false, BhF, (short)0, C[j], false, false);
                }
            }
            __syncthreads();                 // all LDS reads of R done
            if (s == klen - 1) break;        // last step: write out from regs below
            // Split C back into hi/lo LDS (ISA f32 C/D layout: row = v + 8*laneHi)
#pragma unroll
            for (int j = 0; j < 8; ++j) {
                const int col = (cg * 8 + j) * 16 + mrow;
#pragma unroll
                for (int v = 0; v < 8; ++v) {
                    const int row = ti * 16 + laneHi * 8 + v;
                    float x = C[j][v];
                    _Float16 hi = (_Float16)x;
                    Rh[row * LDS_STRIDE + col] = hi;
                    Rl[row * LDS_STRIDE + col] = (_Float16)(x - (float)hi);
                }
            }
            __syncthreads();
        }
        // Final result straight from f32 accumulators (no recombination loss)
#pragma unroll
        for (int j = 0; j < 8; ++j) {
            const int col = (cg * 8 + j) * 16 + mrow;
#pragma unroll
            for (int v = 0; v < 8; ++v) {
                const int row = ti * 16 + laneHi * 8 + v;
                outp[(row << 8) + col] = C[j][v];
            }
        }
    } else {
        // klen == 1: result is M[b0] itself; recombine hi+lo from LDS
        for (int idx = tid; idx < DD; idx += 1024) {
            int row = idx >> 8, col = idx & 255;
            outp[idx] = (float)Rh[row * LDS_STRIDE + col] + (float)Rl[row * LDS_STRIDE + col];
        }
    }
}

// ------------------------------- launcher -----------------------------------

extern "C" void kernel_launch(void* const* d_in, const int* in_sizes, int n_in,
                              void* d_out, int out_size, void* d_ws, size_t ws_size,
                              hipStream_t stream) {
    const float* prim = (const float*)d_in[0];       // (2,256,256)
    // d_in[1] = identity (eye) -- mathematically a no-op here
    const int* unique = (const int*)d_in[2];         // (2048,)
    float* out = (float*)d_out;                      // (n,256,256) f32
    const int n = in_sizes[2];

    // Workspace layout (needs ~3.5 MB)
    float* A  = (float*)d_ws;            // 2*DD
    float* T1 = A  + 2 * DD;
    float* T2 = T1 + 2 * DD;
    float* E0 = T2 + 2 * DD;
    float* E1 = E0 + 2 * DD;
    _Float16* hb   = (_Float16*)(E1 + 2 * DD);
    _Float16* MhRM = hb;                 // 2*DD halves each
    _Float16* MlRM = MhRM + 2 * DD;
    _Float16* MhPK = MlRM + 2 * DD;
    _Float16* MlPK = MhPK + 2 * DD;

    // P = expm(skew) via scaling (2^-6) + Taylor(10) + 6 squarings
    k_skew<<<512, 256, 0, stream>>>(prim, A);
    k_addI<<<512, 256, 0, stream>>>(A, E0);
    const float* Tin = A; float* Tout = T1;
    for (int kk = 2; kk <= 10; ++kk) {
        k_term<<<dim3(256, 2), 256, 0, stream>>>(Tin, A, Tout, E0, 1.0f / (float)kk);
        Tin = Tout;
        Tout = (Tout == T1) ? T2 : T1;
    }
    float* Ein = E0; float* Eout = E1;
    for (int i = 0; i < 6; ++i) {
        k_sq<<<dim3(256, 2), 256, 0, stream>>>(Ein, Eout);
        float* t = Ein; Ein = Eout; Eout = t;
    }
    // Ein now holds P (after even number of swaps -> E0)
    k_pack<<<512, 256, 0, stream>>>(Ein, MhRM, MlRM, MhPK, MlPK);

    // Chained products: one block per position
    k_chain<<<n, 1024, 0, stream>>>(unique, MhRM, MlRM, MhPK, MlPK, out);
}
